// RegionProposalNetwork1d_10393820856832
// MI455X (gfx1250) — matrix operand order repaired and checked by reading, not compiled, not run on hardware
//
#include <hip/hip_runtime.h>
#include <hip/hip_bf16.h>
#include <math.h>

// ---------------------------------------------------------------------------
// RPN-1D fused pipeline for MI455X (gfx1250, wave32, WMMA).
// Position-major LDS layout: B fragments = contiguous ds_load_b128 pairs,
// D epilogue = one ds_store_b128; A fragments pre-swizzled to f16 by kPackW.
// ---------------------------------------------------------------------------

typedef __attribute__((ext_vector_type(16))) _Float16 v16h;
typedef __attribute__((ext_vector_type(8)))  _Float16 v8h;
typedef __attribute__((ext_vector_type(8)))  float    v8f;

#define L_SEQ    131072
#define TILE     64
#define HALO     8
#define TW       80            // TILE + 2*HALO, = 5 WMMA column tiles
#define CBB      128
#define CP       136           // padded channel stride (halfs): 272B rows
#define RPNC     16
#define NA       8
#define NBOX     (L_SEQ * NA)  // 1,048,576 anchors
#define TOPN     6000
#define CAND_MAX 8192
#define POST_N   300
#define NMS_THR  0.7f
#define BN_EPS   1e-5f

// packed WMMA A-fragment pool: layer1 | layer2 | rpn
#define WF_L1    0
#define WF_L2    16384
#define WF_RPN   32768
#define WF_TOTAL 34816

__constant__ float c_widths[NA] = {2.f, 4.f, 6.f, 8.f, 12.f, 16.f, 24.f, 32.f};

struct RpnParams {
  const float* seq;
  const float* dww[4]; const float* dwb[4];   // depthwise w/b (b1,b2,b3,rpn)
  const float* pww0;   const float* pwb[4];   // layer0 pw weights stay f32
  const float* bng[4]; const float* bnb[4];
  const float* bnm[4]; const float* bnv[4];
  const float* clsw;  const float* clsb;
  const float* bbw;   const float* bbb;
  const _Float16* wfrag;                      // pre-swizzled A fragments
  float* scores; float* boxes; unsigned* keys;
};

// ---- pre-swizzle pointwise weights into WMMA A-fragment order (f16) -------
// frag element t: h=t&15, lane=(t>>4)&31, kt=(t>>9)&3, coTile=t>>11
// A 16x32, this code's contiguous k order (same as B): m = coTile*16+(lane&15),
//                              k = kt*32 + (lane>>4)*16 + h
__global__ void kPackW(const float* __restrict__ w1, const float* __restrict__ w2,
                       const float* __restrict__ wr, _Float16* wfrag)
{
  int t = blockIdx.x * blockDim.x + threadIdx.x;
  if (t >= WF_TOTAL) return;
  const float* W; int local;
  if (t < WF_L2)       { W = w1; local = t; }
  else if (t < WF_RPN) { W = w2; local = t - WF_L2; }
  else                 { W = wr; local = t - WF_RPN; }
  int h      = local & 15;
  int lane   = (local >> 4) & 31;
  int kt     = (local >> 9) & 3;
  int coTile = local >> 11;
  int m = coTile * 16 + (lane & 15);
  int k = kt * 32 + ((lane >> 4) * 16) + h;
  wfrag[t] = (_Float16)W[m * CBB + k];
}

// ---- depthwise 3-tap conv, position-major, two channels per iteration -----
__device__ inline void dwT(const _Float16 (*in)[CP], _Float16 (*out)[CP],
                           const float* w3, const float* bias, int nch, int tid)
{
  const int np = nch >> 1;
  for (int idx = tid; idx < TW * np; idx += 256) {
    int j  = idx / np;
    int c0 = (idx - j * np) * 2;
    int jm = j > 0 ? j - 1 : 0;
    int jp = j < TW - 1 ? j + 1 : TW - 1;    // halo-edge clamp (halo only)
    float t0 = w3[c0*3+0] * (float)in[jm][c0]   + w3[c0*3+1] * (float)in[j][c0]
             + w3[c0*3+2] * (float)in[jp][c0]   + bias[c0];
    float t1 = w3[c0*3+3] * (float)in[jm][c0+1] + w3[c0*3+4] * (float)in[j][c0+1]
             + w3[c0*3+5] * (float)in[jp][c0+1] + bias[c0+1];
    out[j][c0]     = (_Float16)t0;
    out[j][c0+1]   = (_Float16)t1;
  }
}

// ---- pointwise conv band via WMMA ----------------------------------------
// B 32x16 (contiguous k order, same as A): lane n=L&15 holds K=(L>>4)*16+h -> contiguous in
// position-major LDS. D f32: lane n=L&15, vgpr r -> channel coBase+8*(L>>4)+r
// -> contiguous v8h store.
__device__ inline void pw_wmma_band(const _Float16 (*in)[CP], _Float16 (*out)[CP],
                                    const _Float16* __restrict__ wf, int coTile,
                                    int ptBegin, int ptEnd,
                                    const float* pwBias, const float* scale,
                                    const float* shift, int p0, int lane)
{
  v16h a[4];
  #pragma unroll
  for (int kt = 0; kt < 4; ++kt)
    a[kt] = *(const v16h*)(wf + ((coTile * 4 + kt) * 32 + lane) * 16);  // 32B aligned

  const int nn     = lane & 15;
  const int hi     = lane >> 4;
  const int kbHi   = hi * 16;
  const int coBase = coTile * 16;

  for (int pt = ptBegin; pt < ptEnd; ++pt) {
    const int row = pt * 16 + nn;
    v8f acc = {};
    #pragma unroll
    for (int kt = 0; kt < 4; ++kt) {
      const _Float16* src = &in[row][kt * 32 + kbHi];   // 16B aligned
      v8h lo  = *(const v8h*)(src);
      v8h hi8 = *(const v8h*)(src + 8);
      v16h b = __builtin_shufflevector(lo, hi8, 0,1,2,3,4,5,6,7,
                                                8,9,10,11,12,13,14,15);
      acc = __builtin_amdgcn_wmma_f32_16x16x32_f16(false, a[kt], false, b,
                                                   (short)0, acc, false, false);
    }
    const int gp  = p0 + pt * 16 + nn;
    const bool inr = (gp >= 0) && (gp < L_SEQ);
    v8h r;
    #pragma unroll
    for (int i = 0; i < 8; ++i) {
      int c = coBase + 8 * hi + i;
      float v = acc[i] + pwBias[c];
      v = fmaxf(v, 0.f) * scale[c] + shift[c];          // ReLU then eval-BN
      r[i] = inr ? (_Float16)v : (_Float16)0.f;         // zero-pad semantics
    }
    *(v8h*)&out[row][coBase + 8 * hi] = r;              // one ds_store_b128
  }
}

// ---------------------------------------------------------------------------
__global__ __launch_bounds__(256)
void kFusedRpn(RpnParams P)
{
  __shared__ _Float16 tA[TW][CP];     // 21.25 KB, position-major
  __shared__ _Float16 tB[TW][CP];     // 21.25 KB
  __shared__ float sScale[4][CBB];
  __shared__ float sShift[4][CBB];
  __shared__ float sPwB[4][CBB];
  __shared__ float sClsW[NA * RPNC];
  __shared__ float sClsB[NA];
  __shared__ float sBbW[2 * NA * RPNC];
  __shared__ float sBbB[2 * NA];

  const int tid  = threadIdx.x;
  const int wave = tid >> 5;
  const int lane = tid & 31;
  const int p0   = (int)blockIdx.x * TILE - HALO;

  // fold BN into scale/shift; stage pw biases + head weights in LDS
  for (int idx = tid; idx < 4 * CBB; idx += 256) {
    int ly = idx >> 7, c = idx & 127;
    int cout = (ly == 3) ? RPNC : CBB;
    if (c < cout) {
      float s = P.bng[ly][c] * rsqrtf(P.bnv[ly][c] + BN_EPS);
      sScale[ly][c] = s;
      sShift[ly][c] = P.bnb[ly][c] - P.bnm[ly][c] * s;
      sPwB[ly][c]   = P.pwb[ly][c];
    }
  }
  for (int idx = tid; idx < NA * RPNC; idx += 256)     sClsW[idx] = P.clsw[idx];
  for (int idx = tid; idx < 2 * NA * RPNC; idx += 256) sBbW[idx]  = P.bbw[idx];
  if (tid < NA) sClsB[tid] = P.clsb[tid];
  else if (tid >= 32 && tid < 32 + 2 * NA) sBbB[tid - 32] = P.bbb[tid - 32];

  // input: 6 channels x TW, zero outside the sequence (conv zero padding)
  for (int idx = tid; idx < 6 * TW; idx += 256) {
    int j = idx / 6, c = idx - j * 6;
    int gp = p0 + j;
    float v = (gp >= 0 && gp < L_SEQ) ? P.seq[c * L_SEQ + gp] : 0.f;
    tA[j][c] = (_Float16)v;
  }
  __syncthreads();

  // ---- layer 0: 6 -> 128 (dw + scalar pw: K=6 too small for WMMA) --------
  dwT(tA, tB, P.dww[0], P.dwb[0], 6, tid);
  __syncthreads();
  for (int idx = tid; idx < TW * 64; idx += 256) {
    int j = idx >> 6, c0 = (idx & 63) * 2;
    float f0 = (float)tB[j][0], f1 = (float)tB[j][1], f2 = (float)tB[j][2];
    float f3 = (float)tB[j][3], f4 = (float)tB[j][4], f5 = (float)tB[j][5];
    int gp = p0 + j;
    bool inr = (gp >= 0) && (gp < L_SEQ);
    #pragma unroll
    for (int q = 0; q < 2; ++q) {
      int co = c0 + q;
      const float* w = P.pww0 + co * 6;
      float acc = sPwB[0][co] + w[0]*f0 + w[1]*f1 + w[2]*f2 + w[3]*f3 + w[4]*f4 + w[5]*f5;
      float v = fmaxf(acc, 0.f) * sScale[0][co] + sShift[0][co];
      tA[j][co] = inr ? (_Float16)v : (_Float16)0.f;
    }
  }
  __syncthreads();

  // ---- layers 1,2: 128 -> 128, WMMA pointwise (8 waves x 5 tiles x 4 K) ---
  dwT(tA, tB, P.dww[1], P.dwb[1], CBB, tid);
  __syncthreads();
  pw_wmma_band(tB, tA, P.wfrag + WF_L1, wave, 0, 5,
               sPwB[1], sScale[1], sShift[1], p0, lane);
  __syncthreads();

  dwT(tA, tB, P.dww[2], P.dwb[2], CBB, tid);
  __syncthreads();
  pw_wmma_band(tB, tA, P.wfrag + WF_L2, wave, 0, 5,
               sPwB[2], sScale[2], sShift[2], p0, lane);
  __syncthreads();

  // ---- RPN layer: 128 -> 16 (waves 0..4 each own one position tile) ------
  dwT(tA, tB, P.dww[3], P.dwb[3], CBB, tid);
  __syncthreads();
  if (wave < 5)
    pw_wmma_band(tB, tA, P.wfrag + WF_RPN, 0, wave, wave + 1,
                 sPwB[3], sScale[3], sShift[3], p0, lane);
  __syncthreads();

  // ---- heads + anchor decode for the 64 central positions -----------------
  for (int item = tid; item < TILE * NA; item += 256) {
    int pl = item >> 3, a = item & 7;
    int j  = HALO + pl;
    int gp = p0 + j;                       // == blockIdx.x*TILE + pl, in [0,L)
    float f[RPNC];
    #pragma unroll
    for (int ci = 0; ci < RPNC; ++ci) f[ci] = (float)tA[j][ci];  // contiguous
    float lg = sClsB[a];
    float d0 = sBbB[2 * a], d1 = sBbB[2 * a + 1];
    #pragma unroll
    for (int ci = 0; ci < RPNC; ++ci) {
      lg += sClsW[a * RPNC + ci] * f[ci];
      d0 += sBbW[(2 * a) * RPNC + ci] * f[ci];
      d1 += sBbW[(2 * a + 1) * RPNC + ci] * f[ci];
    }
    float score = 1.f / (1.f + __expf(-lg));
    float aw   = c_widths[a];
    float pctr = d0 * aw + (float)gp;      // anchor center == position (stride 1)
    float pwd  = __expf(d1) * aw;
    float bs = fminf(fmaxf(pctr - 0.5f * (pwd - 1.f), 0.f), (float)(L_SEQ - 1));
    float be = fminf(fmaxf(pctr + 0.5f * (pwd - 1.f), 0.f), (float)(L_SEQ - 1));
    int gi = gp * NA + a;
    P.scores[gi]        = score;
    P.boxes[2 * gi]     = bs;
    P.boxes[2 * gi + 1] = be;
    unsigned u = __float_as_uint(lg);      // rank by logit (monotone w/ sigmoid)
    P.keys[gi] = (u & 0x80000000u) ? ~u : (u | 0x80000000u);
  }
}

// ---------------------------------------------------------------------------
// Top-6000 threshold via deterministic 2-level radix histogram, then compact.
// ---------------------------------------------------------------------------
__global__ void kZero(unsigned* p, int n)
{
  int i = blockIdx.x * blockDim.x + threadIdx.x;
  if (i < n) p[i] = 0u;
}

__global__ void kHist1(const unsigned* __restrict__ keys, unsigned* hist)
{
  for (int i = blockIdx.x * blockDim.x + threadIdx.x; i < NBOX;
       i += gridDim.x * blockDim.x)
    atomicAdd(&hist[keys[i] >> 20], 1u);
}

__global__ void kPick1(const unsigned* __restrict__ hist, unsigned* meta)
{
  unsigned acc = 0; int t1 = 0;
  for (int b = 4095; b >= 0; --b) {
    unsigned nacc = acc + hist[b];
    if (nacc >= (unsigned)TOPN) { t1 = b; break; }
    acc = nacc;
  }
  meta[0] = (unsigned)t1;   // boundary bin (top 12 key bits)
  meta[1] = acc;            // strictly-above count (< TOPN)
}

__global__ void kHist2(const unsigned* __restrict__ keys,
                       const unsigned* __restrict__ meta, unsigned* hist2)
{
  unsigned t1 = meta[0];
  for (int i = blockIdx.x * blockDim.x + threadIdx.x; i < NBOX;
       i += gridDim.x * blockDim.x) {
    unsigned k = keys[i];
    if ((k >> 20) == t1) atomicAdd(&hist2[(k >> 8) & 0xFFFu], 1u);
  }
}

__global__ void kPick2(const unsigned* __restrict__ hist2, unsigned* meta)
{
  unsigned quota = (unsigned)TOPN - meta[1];  // >= 1 by construction
  unsigned acc = 0; int t2 = 0;
  for (int b = 4095; b >= 0; --b) {
    unsigned nacc = acc + hist2[b];
    if (nacc >= quota) { t2 = b; break; }
    acc = nacc;
  }
  meta[2] = (meta[0] << 20) | ((unsigned)t2 << 8);  // acceptance threshold key
  meta[3] = 0u;                                     // candidate counter
}

__global__ void kCompact(const unsigned* __restrict__ keys,
                         const float* __restrict__ scores,
                         const float* __restrict__ boxes,
                         unsigned* meta, float* cscore, float* cbox)
{
  unsigned T = meta[2];
  for (int i = blockIdx.x * blockDim.x + threadIdx.x; i < NBOX;
       i += gridDim.x * blockDim.x) {
    if (keys[i] >= T) {
      unsigned slot = atomicAdd(&meta[3], 1u);
      if (slot < CAND_MAX) {
        cscore[slot]       = scores[i];
        cbox[2 * slot]     = boxes[2 * i];
        cbox[2 * slot + 1] = boxes[2 * i + 1];
      }
    }
  }
}

// ---------------------------------------------------------------------------
// Greedy NMS: 300 rounds of (argmax over remaining, suppress IoU > 0.7).
// ---------------------------------------------------------------------------
__global__ __launch_bounds__(1024)
void kNms(const unsigned* __restrict__ meta,
          const float* __restrict__ cscore,
          const float* __restrict__ cbox, float* out)
{
  __shared__ float sc[CAND_MAX];        // 32 KB
  __shared__ float redV[32];
  __shared__ int   redI[32];
  __shared__ float bBox[2];
  __shared__ int   bFlag;

  const int tid  = threadIdx.x;
  const int lane = tid & 31;
  const int wv   = tid >> 5;
  int n = (int)meta[3];
  if (n > CAND_MAX) n = CAND_MAX;

  for (int i = tid; i < CAND_MAX; i += 1024)
    sc[i] = (i < n) ? cscore[i] : -INFINITY;
  __syncthreads();

  for (int it = 0; it < POST_N; ++it) {
    float bv = -INFINITY; int bi = 0x7fffffff;
    for (int i = tid; i < n; i += 1024) {
      float v = sc[i];
      if (v > bv || (v == bv && i < bi)) { bv = v; bi = i; }
    }
    #pragma unroll
    for (int off = 16; off > 0; off >>= 1) {
      float ov = __shfl_down(bv, (unsigned)off, 32);
      int   oi = __shfl_down(bi, (unsigned)off, 32);
      if (ov > bv || (ov == bv && oi < bi)) { bv = ov; bi = oi; }
    }
    if (lane == 0) { redV[wv] = bv; redI[wv] = bi; }
    __syncthreads();
    if (tid == 0) {
      float fv = redV[0]; int fi = redI[0];
      for (int w = 1; w < 32; ++w)
        if (redV[w] > fv || (redV[w] == fv && redI[w] < fi)) { fv = redV[w]; fi = redI[w]; }
      if (fi < n && fv > -INFINITY) {
        float s = cbox[2 * fi], e = cbox[2 * fi + 1];
        out[3 * it] = fv; out[3 * it + 1] = s; out[3 * it + 2] = e;
        sc[fi] = -INFINITY;
        bBox[0] = s; bBox[1] = e; bFlag = 1;
      } else {
        out[3 * it] = 0.f; out[3 * it + 1] = 0.f; out[3 * it + 2] = 0.f;
        bFlag = 0;
      }
    }
    __syncthreads();
    if (bFlag) {
      float bs = bBox[0], be = bBox[1];
      float areaJ = be - bs + 1.f;
      for (int i = tid; i < n; i += 1024) {
        if (sc[i] > -INFINITY) {
          float s = cbox[2 * i], e = cbox[2 * i + 1];
          float inter = fmaxf(fminf(e, be) - fmaxf(s, bs) + 1.f, 0.f);
          float iou = inter / ((e - s + 1.f) + areaJ - inter);
          if (iou > NMS_THR) sc[i] = -INFINITY;
        }
      }
    }
    __syncthreads();
  }
}

// ---------------------------------------------------------------------------
extern "C" void kernel_launch(void* const* d_in, const int* in_sizes, int n_in,
                              void* d_out, int out_size, void* d_ws, size_t ws_size,
                              hipStream_t stream)
{
  (void)in_sizes; (void)n_in; (void)out_size; (void)ws_size;

  // setup_inputs() insertion order:
  // 0 sequence; backbone[b] (b=0..2): 1+8b..8+8b = dw_w,dw_b,pw_w,pw_b,bn_g,bn_b,bn_m,bn_v
  // rpn: 25..32 (same order); 33 cls_w, 34 cls_b, 35 bbox_w, 36 bbox_b
  RpnParams P;
  P.seq = (const float*)d_in[0];
  const float* pww[4];
  for (int b = 0; b < 4; ++b) {
    int base = (b < 3) ? (1 + b * 8) : 25;
    P.dww[b] = (const float*)d_in[base + 0];
    P.dwb[b] = (const float*)d_in[base + 1];
    pww[b]   = (const float*)d_in[base + 2];
    P.pwb[b] = (const float*)d_in[base + 3];
    P.bng[b] = (const float*)d_in[base + 4];
    P.bnb[b] = (const float*)d_in[base + 5];
    P.bnm[b] = (const float*)d_in[base + 6];
    P.bnv[b] = (const float*)d_in[base + 7];
  }
  P.pww0 = pww[0];
  P.clsw = (const float*)d_in[33];
  P.clsb = (const float*)d_in[34];
  P.bbw  = (const float*)d_in[35];
  P.bbb  = (const float*)d_in[36];

  // workspace carve-up (~17 MB)
  char* ws = (char*)d_ws;
  float*    scores = (float*)ws;                              // NBOX f32
  float*    boxes  = (float*)(ws + (size_t)NBOX * 4);         // 2*NBOX f32
  unsigned* keys   = (unsigned*)(ws + (size_t)NBOX * 12);     // NBOX u32
  unsigned* hist1  = (unsigned*)(ws + (size_t)NBOX * 16);     // 4096
  unsigned* hist2  = hist1 + 4096;                            // 4096
  unsigned* meta   = hist2 + 4096;                            // 16
  float*    cscore = (float*)(meta + 16);                     // CAND_MAX
  float*    cbox   = cscore + CAND_MAX;                       // 2*CAND_MAX
  _Float16* wfrag  = (_Float16*)(cbox + 2 * CAND_MAX);        // WF_TOTAL (32B-aligned)
  P.scores = scores; P.boxes = boxes; P.keys = keys; P.wfrag = wfrag;

  kZero<<<(4096 + 4096 + 16 + 255) / 256, 256, 0, stream>>>(hist1, 4096 + 4096 + 16);
  kPackW<<<(WF_TOTAL + 255) / 256, 256, 0, stream>>>(pww[1], pww[2], pww[3], wfrag);
  kFusedRpn<<<L_SEQ / TILE, 256, 0, stream>>>(P);
  kHist1<<<2048, 256, 0, stream>>>(keys, hist1);
  kPick1<<<1, 1, 0, stream>>>(hist1, meta);
  kHist2<<<2048, 256, 0, stream>>>(keys, meta, hist2);
  kPick2<<<1, 1, 0, stream>>>(hist2, meta);
  kCompact<<<2048, 256, 0, stream>>>(keys, scores, boxes, meta, cscore, cbox);
  kNms<<<1, 1024, 0, stream>>>(meta, cscore, cbox, (float*)d_out);
}
